// MVoTTokenProcessor_23759759082041
// MI455X (gfx1250) — matrix operand order repaired and checked by reading, not compiled, not run on hardware
//
#include <hip/hip_runtime.h>
#include <hip/hip_bf16.h>
#include <math.h>

#define H 2048
#define TOK 8192
#define BM 64
#define BN 256
#define BK 32
#define ASTR 40   // padded LDS row stride (elems): 80B, 16B-aligned, conflict-free
#define BSTR 40

typedef __attribute__((ext_vector_type(16))) __bf16 v16bf;
typedef __attribute__((ext_vector_type(8)))  float  v8f;

union Frag { v16bf v; uint4 q[2]; };

#if defined(__has_builtin)
#if __has_builtin(__builtin_amdgcn_global_load_async_to_lds_b128)
#define HAVE_ASYNC_LDS 1
#endif
#endif

#ifdef HAVE_ASYNC_LDS
typedef int v4i_raw __attribute__((vector_size(16)));
typedef __attribute__((address_space(1))) v4i_raw gbl_v4i_t;
typedef __attribute__((address_space(3))) v4i_raw lds_v4i_t;
#endif

// 16-byte global -> LDS copy. Async (ASYNCcnt-tracked, no VGPR round trip)
// when the gfx1250 builtin is available, plain load/store otherwise.
__device__ __forceinline__ void copy16_g2l(const unsigned short* g,
                                           unsigned short* l) {
#ifdef HAVE_ASYNC_LDS
  __builtin_amdgcn_global_load_async_to_lds_b128(
      (gbl_v4i_t*)g, (lds_v4i_t*)l, 0, 0);
#else
  *(uint4*)l = *(const uint4*)g;
#endif
}

__device__ __forceinline__ void wait_async_lds() {
#ifdef HAVE_ASYNC_LDS
  asm volatile("s_wait_asynccnt 0" ::: "memory");
#endif
}

__device__ __forceinline__ unsigned short f2bf(float f) {
  unsigned int u = __float_as_uint(f);
  unsigned int r = u + 0x7FFFu + ((u >> 16) & 1u);   // round-to-nearest-even
  return (unsigned short)(r >> 16);
}

__device__ __forceinline__ float gelu_exact(float x) {
  return 0.5f * x * (1.0f + erff(x * 0.70710678118654752f));
}

// ---------------------------------------------------------------------------
// Weight convert: fp32 [k][n] -> bf16 transposed [n][k], LDS tile transpose.
// grid (64,64,4) block 256
// ---------------------------------------------------------------------------
__global__ __launch_bounds__(256) void wconv_kernel(
    const float* __restrict__ w0, const float* __restrict__ w1,
    const float* __restrict__ w2, const float* __restrict__ w3,
    unsigned short* __restrict__ o0, unsigned short* __restrict__ o1,
    unsigned short* __restrict__ o2, unsigned short* __restrict__ o3) {
  const float* src; unsigned short* dst;
  switch (blockIdx.z) {
    case 0: src = w0; dst = o0; break;
    case 1: src = w1; dst = o1; break;
    case 2: src = w2; dst = o2; break;
    default: src = w3; dst = o3; break;
  }
  __shared__ float tile[32][33];
  int tx = threadIdx.x & 31, ty = threadIdx.x >> 5;     // ty in [0,8)
  int k0 = blockIdx.y * 32, n0 = blockIdx.x * 32;
#pragma unroll
  for (int i = 0; i < 4; ++i)
    tile[ty + i * 8][tx] = src[(size_t)(k0 + ty + i * 8) * H + n0 + tx];
  __syncthreads();
#pragma unroll
  for (int i = 0; i < 4; ++i) {
    int nn = ty + i * 8;
    dst[(size_t)(n0 + nn) * H + k0 + tx] = f2bf(tile[tx][nn]);
  }
}

// ---------------------------------------------------------------------------
// LayerNorm + bf16 convert. grid 8192, block 256, 8 elems/thread.
// ---------------------------------------------------------------------------
__global__ __launch_bounds__(256) void ln_kernel(
    const float* __restrict__ x, const float* __restrict__ g,
    const float* __restrict__ b, unsigned short* __restrict__ xnb) {
  int row = blockIdx.x;
  int tid = threadIdx.x;
  const float* xr = x + (size_t)row * H;
  float4 v0 = *(const float4*)(xr + tid * 8);
  float4 v1 = *(const float4*)(xr + tid * 8 + 4);
  float s  = v0.x + v0.y + v0.z + v0.w + v1.x + v1.y + v1.z + v1.w;
  float ss = v0.x*v0.x + v0.y*v0.y + v0.z*v0.z + v0.w*v0.w
           + v1.x*v1.x + v1.y*v1.y + v1.z*v1.z + v1.w*v1.w;
  __shared__ float red[256], red2[256];
  red[tid] = s; red2[tid] = ss;
  __syncthreads();
  for (int off = 128; off; off >>= 1) {
    if (tid < off) { red[tid] += red[tid + off]; red2[tid] += red2[tid + off]; }
    __syncthreads();
  }
  float mean = red[0] * (1.0f / H);
  float var  = red2[0] * (1.0f / H) - mean * mean;
  float inv  = rsqrtf(var + 1e-12f);
  float4 g0 = *(const float4*)(g + tid * 8);
  float4 g1 = *(const float4*)(g + tid * 8 + 4);
  float4 b0 = *(const float4*)(b + tid * 8);
  float4 b1 = *(const float4*)(b + tid * 8 + 4);
  float o[8];
  o[0] = (v0.x - mean) * inv * g0.x + b0.x;
  o[1] = (v0.y - mean) * inv * g0.y + b0.y;
  o[2] = (v0.z - mean) * inv * g0.z + b0.z;
  o[3] = (v0.w - mean) * inv * g0.w + b0.w;
  o[4] = (v1.x - mean) * inv * g1.x + b1.x;
  o[5] = (v1.y - mean) * inv * g1.y + b1.y;
  o[6] = (v1.z - mean) * inv * g1.z + b1.z;
  o[7] = (v1.w - mean) * inv * g1.w + b1.w;
  uint4 pk;
  pk.x = (unsigned)f2bf(o[0]) | ((unsigned)f2bf(o[1]) << 16);
  pk.y = (unsigned)f2bf(o[2]) | ((unsigned)f2bf(o[3]) << 16);
  pk.z = (unsigned)f2bf(o[4]) | ((unsigned)f2bf(o[5]) << 16);
  pk.w = (unsigned)f2bf(o[6]) | ((unsigned)f2bf(o[7]) << 16);
  *(uint4*)(xnb + (size_t)row * H + tid * 8) = pk;
}

// ---------------------------------------------------------------------------
// Token compaction by type (deterministic single-block scan).
// gather[0..8191]=text token ids, gather[8192..]=image token ids.
// ---------------------------------------------------------------------------
__global__ __launch_bounds__(256) void scan_kernel(
    const int* __restrict__ tt, int* __restrict__ gather,
    int* __restrict__ counts) {
  int tid = threadIdx.x;
  int base = tid * 32;
  int c = 0;
#pragma unroll 4
  for (int i = 0; i < 32; ++i) c += (tt[base + i] == 0) ? 1 : 0;
  __shared__ int sc[256];
  sc[tid] = c;
  __syncthreads();
  for (int off = 1; off < 256; off <<= 1) {           // inclusive scan
    int v = sc[tid];
    int add = (tid >= off) ? sc[tid - off] : 0;
    __syncthreads();
    sc[tid] = v + add;
    __syncthreads();
  }
  int tr = (tid == 0) ? 0 : sc[tid - 1];              // text rank before chunk
  if (tid == 255) { counts[0] = sc[255]; counts[1] = TOK - sc[255]; }
  for (int i = 0; i < 32; ++i) {
    int idx = base + i;
    if (tt[idx] == 0) { gather[tr] = idx; ++tr; }
    else               { gather[TOK + (idx - tr)] = idx; }
  }
}

// ---------------------------------------------------------------------------
// Gathered GEMM, bf16 WMMA 16x16x32, fp32 accumulate, double-buffered LDS
// with async global->LDS staging.
// MODE 0: out_bf16 = GELU(A @ Wt + bias)   (A = xn bf16)
// MODE 1: out_f32  = resid + A @ Wt + bias (A = h bf16)
// Block: 256 thr = 8 waves (2x4), block tile 64x256, wave tile 32x64, BK=32.
// grid (H/BN=8, TOK/BM=128, 2=type)
// ---------------------------------------------------------------------------
template <int MODE>
__global__ __launch_bounds__(256) void gemm_kernel(
    const unsigned short* __restrict__ Abf,
    const unsigned short* __restrict__ Wt_text,
    const unsigned short* __restrict__ Wt_image,
    const float* __restrict__ bias_text, const float* __restrict__ bias_image,
    const int* __restrict__ gather, const int* __restrict__ counts,
    const float* __restrict__ resid,
    unsigned short* __restrict__ outb, float* __restrict__ outf) {
  const int type = blockIdx.z;
  const int cnt = counts[type];
  const int rowBase = blockIdx.y * BM;
  if (rowBase >= cnt) return;                          // compacted early-exit
  const unsigned short* __restrict__ W = type ? Wt_image : Wt_text;
  const float* __restrict__ bias = type ? bias_image : bias_text;
  const int* __restrict__ gat = gather + type * TOK;
  const int n0 = blockIdx.x * BN;

  const int tid  = threadIdx.x;
  const int lane = tid & 31;
  const int wid  = tid >> 5;
  const int waveM = wid >> 2;                          // 0..1
  const int waveN = wid & 3;                           // 0..3
  const int half = lane >> 4;                          // K-half select
  const int lm   = lane & 15;

  __shared__ unsigned short As[2][BM * ASTR];
  __shared__ unsigned short Bs[2][BN * BSTR];

  // A staging: thread -> row tid>>2 (0..63), 8-elem chunk (tid&3)*8 (16B).
  const int aRow = tid >> 2;
  const int aChunk = (tid & 3) * 8;
  int lr = rowBase + aRow;
  int gRowA = gat[lr < cnt ? lr : (cnt - 1)];          // clamp pad rows
  const unsigned short* __restrict__ aPtr = Abf + (size_t)gRowA * H + aChunk;
  // B staging: thread t stages 32 bf16 (4x16B) of transposed-weight row n0+t.
  const unsigned short* __restrict__ bPtr = W + (size_t)(n0 + tid) * H;

  v8f cacc[2][4] = {};

  // A frag (ISA 7.12.2, 16-bit A 16x32): lane half -> K groups {h*8, 16+h*8}
  const int aBase0 = (waveM * 32 + lm) * ASTR + half * 8;   // m-sub 0
  const int aBase1 = aBase0 + 16 * ASTR;                    // m-sub 1

  // stage K-tile k0 into LDS buffer `buf`
  auto stage = [&](int k0, int buf) {
    copy16_g2l(aPtr + k0, &As[buf][aRow * ASTR + aChunk]);
#pragma unroll
    for (int c = 0; c < 4; ++c)
      copy16_g2l(bPtr + k0 + c * 8, &Bs[buf][tid * BSTR + c * 8]);
  };

  stage(0, 0);
  wait_async_lds();
  __syncthreads();

  for (int k0 = 0; k0 < H; k0 += BK) {
    const int cur = (k0 >> 5) & 1;
    if (k0 + BK < H) {
      stage(k0 + BK, cur ^ 1);                         // async prefetch next
      if (k0 + 2 * BK < H) __builtin_prefetch(bPtr + k0 + 2 * BK, 0, 1);
    }
    Frag fa0, fa1;
    fa0.q[0] = *(const uint4*)&As[cur][aBase0];
    fa0.q[1] = *(const uint4*)&As[cur][aBase0 + 16];
    fa1.q[0] = *(const uint4*)&As[cur][aBase1];
    fa1.q[1] = *(const uint4*)&As[cur][aBase1 + 16];
#pragma unroll
    for (int t = 0; t < 4; ++t) {
      // B frag (n-major in LDS): lane half -> K groups {h*8, 16+h*8}, matching A.
      const int n = waveN * 64 + t * 16 + lm;
      Frag fb;
      fb.q[0] = *(const uint4*)&Bs[cur][n * BSTR + half * 8];
      fb.q[1] = *(const uint4*)&Bs[cur][n * BSTR + half * 8 + 16];
      cacc[0][t] = __builtin_amdgcn_wmma_f32_16x16x32_bf16(
          false, fa0.v, false, fb.v, (short)0, cacc[0][t], false, false);
      cacc[1][t] = __builtin_amdgcn_wmma_f32_16x16x32_bf16(
          false, fa1.v, false, fb.v, (short)0, cacc[1][t], false, false);
    }
    wait_async_lds();
    __syncthreads();
  }

  // C layout (ISA 7.12.2): elem (m,n): m = v + 8*half, n = lane&15.
  int tok[2][8]; bool valid[2][8];
#pragma unroll
  for (int ms = 0; ms < 2; ++ms) {
#pragma unroll
    for (int v = 0; v < 8; ++v) {
      int r = rowBase + waveM * 32 + ms * 16 + (half << 3) + v;
      valid[ms][v] = (r < cnt);
      tok[ms][v] = gat[valid[ms][v] ? r : (cnt - 1)];
    }
  }
#pragma unroll
  for (int t = 0; t < 4; ++t) {
    const int col = n0 + waveN * 64 + t * 16 + lm;
    const float bv = bias[col];
#pragma unroll
    for (int ms = 0; ms < 2; ++ms) {
#pragma unroll
      for (int v = 0; v < 8; ++v) {
        if (!valid[ms][v]) continue;
        float acc = cacc[ms][t][v] + bv;
        if (MODE == 0) {
          outb[(size_t)tok[ms][v] * H + col] = f2bf(gelu_exact(acc));
        } else {
          outf[(size_t)tok[ms][v] * H + col] =
              resid[(size_t)tok[ms][v] * H + col] + acc;
        }
      }
    }
  }
}

// ---------------------------------------------------------------------------
extern "C" void kernel_launch(void* const* d_in, const int* in_sizes, int n_in,
                              void* d_out, int out_size, void* d_ws,
                              size_t ws_size, hipStream_t stream) {
  const float* hidden = (const float*)d_in[0];
  const int*   ttids  = (const int*)d_in[1];
  const float* gamma  = (const float*)d_in[2];
  const float* beta   = (const float*)d_in[3];
  const float* tw1 = (const float*)d_in[4];
  const float* tb1 = (const float*)d_in[5];
  const float* tw2 = (const float*)d_in[6];
  const float* tb2 = (const float*)d_in[7];
  const float* iw1 = (const float*)d_in[8];
  const float* ib1 = (const float*)d_in[9];
  const float* iw2 = (const float*)d_in[10];
  const float* ib2 = (const float*)d_in[11];
  float* out = (float*)d_out;

  char* ws = (char*)d_ws;
  const size_t ACT = (size_t)TOK * H * 2;   // 32 MB
  const size_t WSZ = (size_t)H * H * 2;     // 8 MB
  unsigned short* xnb  = (unsigned short*)(ws);
  unsigned short* hb   = (unsigned short*)(ws + ACT);
  unsigned short* twb1 = (unsigned short*)(ws + 2 * ACT);
  unsigned short* twb2 = (unsigned short*)(ws + 2 * ACT + WSZ);
  unsigned short* iwb1 = (unsigned short*)(ws + 2 * ACT + 2 * WSZ);
  unsigned short* iwb2 = (unsigned short*)(ws + 2 * ACT + 3 * WSZ);
  int* gather = (int*)(ws + 2 * ACT + 4 * WSZ);
  int* counts = (int*)(ws + 2 * ACT + 4 * WSZ + (size_t)2 * TOK * 4);

  wconv_kernel<<<dim3(H / 32, H / 32, 4), 256, 0, stream>>>(
      tw1, tw2, iw1, iw2, twb1, twb2, iwb1, iwb2);
  ln_kernel<<<TOK, 256, 0, stream>>>(hidden, gamma, beta, xnb);
  scan_kernel<<<1, 256, 0, stream>>>(ttids, gather, counts);
  gemm_kernel<0><<<dim3(H / BN, TOK / BM, 2), 256, 0, stream>>>(
      xnb, twb1, iwb1, tb1, ib1, gather, counts, nullptr, hb, nullptr);
  gemm_kernel<1><<<dim3(H / BN, TOK / BM, 2), 256, 0, stream>>>(
      hb, twb2, iwb2, tb2, ib2, gather, counts, hidden, nullptr, out);
}
